// HoloMambaRec_76390288326723
// MI455X (gfx1250) — matrix operand order repaired and checked by reading, not compiled, not run on hardware
//
#include <hip/hip_runtime.h>

typedef __bf16 bf16;
typedef __attribute__((ext_vector_type(16))) __bf16 v16bf;
typedef __attribute__((ext_vector_type(8)))  __bf16 v8bf;
typedef __attribute__((ext_vector_type(8)))  float  v8f;

#define BATCH   16
#define SEQL    256
#define ROWS    (BATCH * SEQL)   // 4096
#define DM      256
#define DI      512
#define DSTATE  16
#define DTR     16
#define NL      4
#define NITEMS  30000
#define NITEMS_PAD 30016         // 469 * 64

// ---------------------------------------------------------------------------
// tiled transpose + f32->bf16: out[Npad][K] = in[K][N]^T, zero-fill n >= N.
// block dim3(32,8); grid (K/32, Npad/32). K, Npad multiples of 32.
// ---------------------------------------------------------------------------
__global__ __launch_bounds__(256)
void hm_transpose_to_bf16(const float* __restrict__ in, bf16* __restrict__ out,
                          int K, int N) {
    __shared__ float tile[32][33];
    const int tx = threadIdx.x, ty = threadIdx.y;
    const int k0 = blockIdx.x * 32, n0 = blockIdx.y * 32;
    #pragma unroll
    for (int i = 0; i < 4; ++i) {
        const int k = k0 + ty + i * 8;
        const int n = n0 + tx;
        tile[ty + i * 8][tx] = (n < N) ? in[(size_t)k * N + n] : 0.0f;
    }
    __syncthreads();
    #pragma unroll
    for (int i = 0; i < 4; ++i) {
        const int n = n0 + ty + i * 8;
        out[(size_t)n * K + (k0 + tx)] = (bf16)tile[tx][ty + i * 8];
    }
}

// ---------------------------------------------------------------------------
// block-wide sum over 256 threads
// ---------------------------------------------------------------------------
__device__ __forceinline__ float hm_block_sum256(float v, float* red, int k) {
    red[k] = v;
    __syncthreads();
    #pragma unroll
    for (int s = 128; s > 0; s >>= 1) {
        if (k < s) red[k] += red[k + s];
        __syncthreads();
    }
    float r = red[0];
    __syncthreads();
    return r;
}

// ---------------------------------------------------------------------------
// embedding gather + holographic bind (circular conv / 16) + layernorm
// ---------------------------------------------------------------------------
__global__ __launch_bounds__(256)
void hm_embed_bind_ln(const int* __restrict__ seq, const int* __restrict__ attr,
                      const float* __restrict__ item_emb, const float* __restrict__ attr_emb,
                      const float* __restrict__ alpha_p,
                      const float* __restrict__ ln_s, const float* __restrict__ ln_b,
                      float* __restrict__ x) {
    const int row = blockIdx.x;
    const int k   = threadIdx.x;
    __shared__ float ei[DM], ea[DM], red[256];
    const int it = seq[row], at = attr[row];
    ei[k] = (it == 0) ? 0.0f : item_emb[(size_t)it * DM + k];
    ea[k] = (at == 0) ? 0.0f : attr_emb[(size_t)at * DM + k];
    __syncthreads();
    float conv = 0.0f;
    #pragma unroll 8
    for (int j = 0; j < DM; ++j) conv += ei[j] * ea[(k - j) & (DM - 1)];
    // ortho rfft/irfft pair => circconv / sqrt(256)
    float v = ei[k] + alpha_p[0] * conv * 0.0625f;
    float mean = hm_block_sum256(v, red, k) * (1.0f / DM);
    float dv = v - mean;
    float var = hm_block_sum256(dv * dv, red, k) * (1.0f / DM);
    x[(size_t)row * DM + k] = dv * rsqrtf(var + 1e-5f) * ln_s[k] + ln_b[k];
}

// ---------------------------------------------------------------------------
// per-layer layernorm, emits bf16 activations for WMMA
// ---------------------------------------------------------------------------
__global__ __launch_bounds__(256)
void hm_layernorm_bf16(const float* __restrict__ x, const float* __restrict__ s,
                       const float* __restrict__ b, bf16* __restrict__ out) {
    const int row = blockIdx.x;
    const int k   = threadIdx.x;
    __shared__ float red[256];
    float v = x[(size_t)row * DM + k];
    float mean = hm_block_sum256(v, red, k) * (1.0f / DM);
    float dv = v - mean;
    float var = hm_block_sum256(dv * dv, red, k) * (1.0f / DM);
    out[(size_t)row * DM + k] = (bf16)(dv * rsqrtf(var + 1e-5f) * s[k] + b[k]);
}

// ---------------------------------------------------------------------------
// bf16 WMMA GEMM: C[M,N] = A[M,K] @ Bt^T (+resid), optional bf16 mirror.
// Bt is pre-transposed [Npad][K] bf16 with Npad = 64-multiple, zero-filled,
// so the hot loop has NO bounds checks, NO LDS, NO barriers:
//   per K-step/lane: 2x b128 (A frag) + 4x 2x b128 (B frags) + 4x WMMA.
// All B traffic is L2-resident (weights < 16 MB bf16, global L2 = 192 MB).
// Fragment layouts per CDNA5 ISA 7.12.2:
//   A 16x32: lane l16 = row, half 0 -> K {0..7,16..23}, half 1 -> {8..15,24..31}
//   B 32x16: lane l16 = col, half 0 -> K {0..7,16..23}, half 1 -> {8..15,24..31}
//   C/D:     VGPR r -> M = 8*half + r, N = l16
// ---------------------------------------------------------------------------
__global__ __launch_bounds__(256)
void hm_gemm_bf16_wmma(const bf16* __restrict__ A, const bf16* __restrict__ Bt,
                       const float* __restrict__ resid,
                       float* __restrict__ C, bf16* __restrict__ Cbf,
                       int M, int N, int K) {
    const int tid  = threadIdx.x;
    const int wave = tid >> 5;
    const int lane = tid & 31;
    const int hv   = lane >> 4;
    const int l16  = lane & 15;
    const int m0   = blockIdx.y * 128 + wave * 16;
    const int n0   = blockIdx.x * 64;
    (void)M;

    v8f acc[4] = {};
    const bf16* ap  = A  + (size_t)(m0 + l16) * K + (hv << 3);
    const bf16* bp0 = Bt + (size_t)(n0 + l16) * K + (hv << 3);

    for (int kk = 0; kk < K; kk += 32) {
        v8bf alo = *(const v8bf*)(ap);
        v8bf ahi = *(const v8bf*)(ap + 16);
        v16bf afrag = __builtin_shufflevector(alo, ahi, 0, 1, 2, 3, 4, 5, 6, 7,
                                              8, 9, 10, 11, 12, 13, 14, 15);
        if (kk + 32 < K) __builtin_prefetch(ap + 32, 0, 1);
        ap += 32;
        #pragma unroll
        for (int j = 0; j < 4; ++j) {
            const bf16* bp = bp0 + (size_t)j * 16 * K;
            v8bf blo = *(const v8bf*)(bp);
            v8bf bhi = *(const v8bf*)(bp + 16);
            v16bf bfrag = __builtin_shufflevector(blo, bhi, 0, 1, 2, 3, 4, 5, 6, 7,
                                                  8, 9, 10, 11, 12, 13, 14, 15);
            acc[j] = __builtin_amdgcn_wmma_f32_16x16x32_bf16(
                false, afrag, false, bfrag, (short)0, acc[j], false, false);
        }
        bp0 += 32;
    }

    const int rbase = m0 + (hv << 3);
    #pragma unroll
    for (int j = 0; j < 4; ++j) {
        const int col = n0 + j * 16 + l16;
        if (col < N) {
            #pragma unroll
            for (int r = 0; r < 8; ++r) {
                const size_t idx = (size_t)(rbase + r) * N + col;
                float v = acc[j][r];
                if (resid) v += resid[idx];
                C[idx] = v;
                if (Cbf) Cbf[idx] = (bf16)v;
            }
        }
    }
}

// ---------------------------------------------------------------------------
// causal depthwise conv (k=4, left pad 3) + bias + SiLU; fp32 + bf16 out
// ---------------------------------------------------------------------------
__global__ __launch_bounds__(256)
void hm_conv_silu(const float* __restrict__ xz, const float* __restrict__ w,
                  const float* __restrict__ wb, float* __restrict__ xc,
                  bf16* __restrict__ xcbf) {
    const int idx = blockIdx.x * 256 + threadIdx.x;      // over ROWS*DI
    const int d   = idx & (DI - 1);
    const int row = idx >> 9;
    const int l   = row & (SEQL - 1);
    const float* wd = w + d * 4;
    float acc = wb[d];
    #pragma unroll
    for (int t = 0; t < 4; ++t) {
        const int ll = l - 3 + t;
        if (ll >= 0) acc += xz[(size_t)(row - 3 + t) * (2 * DI) + d] * wd[t];
    }
    const float o = acc / (1.0f + __expf(-acc));         // silu
    xc[idx]   = o;
    xcbf[idx] = (bf16)o;
}

// ---------------------------------------------------------------------------
// dt = softplus(dt_raw @ dt_w + dt_b)   (K=16, VALU)
// ---------------------------------------------------------------------------
__global__ __launch_bounds__(256)
void hm_dt_softplus(const float* __restrict__ xp, const float* __restrict__ dt_w,
                    const float* __restrict__ dt_b, float* __restrict__ dt) {
    const int idx = blockIdx.x * 256 + threadIdx.x;      // over ROWS*DI
    const int d   = idx & (DI - 1);
    const int row = idx >> 9;
    const float* xr = xp + (size_t)row * 48;
    float s = dt_b[d];
    #pragma unroll
    for (int r = 0; r < DTR; ++r) s += xr[r] * dt_w[r * DI + d];
    dt[idx] = (s > 20.0f) ? s : log1pf(__expf(s));
}

// ---------------------------------------------------------------------------
// selective scan: one thread per (b, d_inner); 16 states in registers.
// fuses y = (scan + xc*D) * silu(z), emits bf16 for out_proj GEMM.
// ---------------------------------------------------------------------------
__global__ __launch_bounds__(256)
void hm_ssm_scan(const float* __restrict__ dt, const float* __restrict__ xp,
                 const float* __restrict__ xc, const float* __restrict__ xz,
                 const float* __restrict__ A_log, const float* __restrict__ Dp,
                 bf16* __restrict__ ypbf) {
    const int idx = blockIdx.x * 256 + threadIdx.x;      // over BATCH*DI = 8192
    const int d = idx & (DI - 1);
    const int b = idx >> 9;
    float A[DSTATE], h[DSTATE];
    #pragma unroll
    for (int n = 0; n < DSTATE; ++n) {
        A[n] = -__expf(A_log[(size_t)d * DSTATE + n]);
        h[n] = 0.0f;
    }
    const float dpv = Dp[d];
    for (int l = 0; l < SEQL; ++l) {
        const int row = b * SEQL + l;
        const float dtv = dt[(size_t)row * DI + d];
        const float xv  = xc[(size_t)row * DI + d];
        const float* Bp = xp + (size_t)row * 48 + 16;
        const float* Cp = xp + (size_t)row * 48 + 32;
        float y = 0.0f;
        #pragma unroll
        for (int n = 0; n < DSTATE; ++n) {
            h[n] = __expf(dtv * A[n]) * h[n] + dtv * Bp[n] * xv;
            y += h[n] * Cp[n];
        }
        const float z = xz[(size_t)row * (2 * DI) + DI + d];
        const float o = (y + xv * dpv) * (z / (1.0f + __expf(-z)));
        ypbf[(size_t)row * DI + d] = (bf16)o;
    }
}

// ---------------------------------------------------------------------------
// host orchestration
// ---------------------------------------------------------------------------
extern "C" void kernel_launch(void* const* d_in, const int* in_sizes, int n_in,
                              void* d_out, int out_size, void* d_ws, size_t ws_size,
                              hipStream_t stream) {
    (void)in_sizes; (void)n_in; (void)out_size; (void)ws_size;
    const int*   seq      = (const int*)d_in[0];
    const int*   attr     = (const int*)d_in[1];
    const float* item_emb = (const float*)d_in[2];
    const float* attr_emb = (const float*)d_in[3];
    const float* alpha    = (const float*)d_in[4];
    const float* ln_s     = (const float*)d_in[5];
    const float* ln_b     = (const float*)d_in[6];
    const float* in_w     = (const float*)d_in[7];
    const float* conv_w   = (const float*)d_in[8];
    const float* conv_b   = (const float*)d_in[9];
    const float* xp_w     = (const float*)d_in[10];
    const float* dt_w     = (const float*)d_in[11];
    const float* dt_b     = (const float*)d_in[12];
    const float* A_log    = (const float*)d_in[13];
    const float* D_p      = (const float*)d_in[14];
    const float* out_w    = (const float*)d_in[15];
    const float* nrm_s    = (const float*)d_in[16];
    const float* nrm_b    = (const float*)d_in[17];
    const float* head_w   = (const float*)d_in[18];

    char* ws = (char*)d_ws;
    size_t off = 0;
    auto take = [&](size_t bytes) -> char* {
        char* p = ws + off;
        off = (off + bytes + 255) & ~(size_t)255;
        return p;
    };
    // pre-transposed bf16 weights:  [Npad][K]
    bf16*  in_w_t   = (bf16*) take((size_t)NL * 1024 * DM * 2);        // Npad=1024, K=256
    bf16*  xp_w_t   = (bf16*) take((size_t)NL * 64 * DI * 2);          // Npad=64,   K=512
    bf16*  out_w_t  = (bf16*) take((size_t)NL * DM * DI * 2);          // Npad=256,  K=512
    bf16*  head_w_t = (bf16*) take((size_t)NITEMS_PAD * DM * 2);       // Npad=30016,K=256
    float* x        = (float*)take((size_t)ROWS * DM * 4);
    bf16*  x_bf     = (bf16*) take((size_t)ROWS * DM * 2);
    bf16*  xln_bf   = (bf16*) take((size_t)ROWS * DM * 2);
    float* xz       = (float*)take((size_t)ROWS * 2 * DI * 4);
    float* xc       = (float*)take((size_t)ROWS * DI * 4);
    bf16*  xc_bf    = (bf16*) take((size_t)ROWS * DI * 2);
    float* xpout    = (float*)take((size_t)ROWS * 48 * 4);
    float* dtbuf    = (float*)take((size_t)ROWS * DI * 4);
    bf16*  yp_bf    = (bf16*) take((size_t)ROWS * DI * 2);

    const dim3 tb(32, 8);
    for (int l = 0; l < NL; ++l) {
        hm_transpose_to_bf16<<<dim3(DM / 32, 1024 / 32), tb, 0, stream>>>(
            in_w + (size_t)l * DM * 1024, in_w_t + (size_t)l * 1024 * DM, DM, 1024);
        hm_transpose_to_bf16<<<dim3(DI / 32, 64 / 32), tb, 0, stream>>>(
            xp_w + (size_t)l * DI * 48, xp_w_t + (size_t)l * 64 * DI, DI, 48);
        hm_transpose_to_bf16<<<dim3(DI / 32, DM / 32), tb, 0, stream>>>(
            out_w + (size_t)l * DI * DM, out_w_t + (size_t)l * DM * DI, DI, DM);
    }
    hm_transpose_to_bf16<<<dim3(DM / 32, NITEMS_PAD / 32), tb, 0, stream>>>(
        head_w, head_w_t, DM, NITEMS);

    hm_embed_bind_ln<<<ROWS, 256, 0, stream>>>(seq, attr, item_emb, attr_emb,
                                               alpha, ln_s, ln_b, x);

    for (int l = 0; l < NL; ++l) {
        hm_layernorm_bf16<<<ROWS, 256, 0, stream>>>(x, nrm_s + l * DM, nrm_b + l * DM, xln_bf);
        hm_gemm_bf16_wmma<<<dim3((2 * DI) / 64, ROWS / 128), 256, 0, stream>>>(
            xln_bf, in_w_t + (size_t)l * 1024 * DM, nullptr, xz, nullptr,
            ROWS, 2 * DI, DM);
        hm_conv_silu<<<(ROWS * DI) / 256, 256, 0, stream>>>(
            xz, conv_w + (size_t)l * DI * 4, conv_b + (size_t)l * DI, xc, xc_bf);
        hm_gemm_bf16_wmma<<<dim3(1, ROWS / 128), 256, 0, stream>>>(
            xc_bf, xp_w_t + (size_t)l * 64 * DI, nullptr, xpout, nullptr,
            ROWS, 48, DI);
        hm_dt_softplus<<<(ROWS * DI) / 256, 256, 0, stream>>>(
            xpout, dt_w + (size_t)l * DTR * DI, dt_b + (size_t)l * DI, dtbuf);
        hm_ssm_scan<<<(BATCH * DI) / 256, 256, 0, stream>>>(
            dtbuf, xpout, xc, xz, A_log + (size_t)l * DI * DSTATE, D_p + (size_t)l * DI, yp_bf);
        hm_gemm_bf16_wmma<<<dim3(DM / 64, ROWS / 128), 256, 0, stream>>>(
            yp_bf, out_w_t + (size_t)l * DM * DI, x, x, x_bf, ROWS, DM, DI);
    }

    hm_gemm_bf16_wmma<<<dim3(NITEMS_PAD / 64, ROWS / 128), 256, 0, stream>>>(
        x_bf, head_w_t, nullptr, (float*)d_out, nullptr, ROWS, NITEMS, DM);
}
